// AngleAwareTripletLoss_80221399155347
// MI455X (gfx1250) — matrix operand order repaired and checked by reading, not compiled, not run on hardware
//
#include <hip/hip_runtime.h>

// ---------------------------------------------------------------------------
// AngleAwareTripletLoss for MI455X (gfx1250, wave32, WMMA).
// B=4096, F=512, A=3.  Gram matrix via V_WMMA_F32_16X16X32_BF16 with fused
// row-argmax/argmin epilogue; no B x B matrix ever materialized.
// Round 2: squared-distance epilogue (no sqrt in O(B^2) path), branchless
// cndmask tracker updates, dual column-tile per wave (A-frag reuse, 2 WMMA
// per A load), vectorized bf16 stores.
// ---------------------------------------------------------------------------

typedef __attribute__((ext_vector_type(16))) __bf16 v16bf;
typedef __attribute__((ext_vector_type(8)))  __bf16 v8bf;
typedef __attribute__((ext_vector_type(8)))  float  v8f;

#define B_DIM   4096
#define F_DIM   512
#define STRIP   16
#define NWAVES  8
#define NTILES  (B_DIM / 16)

__device__ __forceinline__ float wave_reduce_add(float v) {
#pragma unroll
  for (int m = 16; m >= 1; m >>= 1) v += __shfl_xor(v, m, 32);
  return v;
}

// ---------------------------------------------------------------------------
// Kernel 1: per-row norms, bf16 normalized features, per-row recon cosine.
// One wave32 per row; lane handles 16 contiguous elements.
// ---------------------------------------------------------------------------
__global__ void prep_kernel(const float* __restrict__ f,
                            const float* __restrict__ forig,
                            __bf16* __restrict__ fhat,
                            float* __restrict__ recon_row) {
  const int wave = threadIdx.x >> 5;
  const int lane = threadIdx.x & 31;
  const int row  = blockIdx.x * NWAVES + wave;
  if (row >= B_DIM) return;

  const float* fr = f     + (size_t)row * F_DIM + lane * 16;
  const float* fo = forig + (size_t)row * F_DIM + lane * 16;

  float buf[16];
  float ss = 0.f, dot = 0.f, sso = 0.f;
#pragma unroll
  for (int i = 0; i < 16; ++i) {
    float a = fr[i], b = fo[i];
    buf[i] = a;
    ss += a * a;  dot += a * b;  sso += b * b;
  }
  ss  = wave_reduce_add(ss);
  dot = wave_reduce_add(dot);
  sso = wave_reduce_add(sso);

  const float norm = sqrtf(ss);
  const float inv  = (norm > 1e-20f) ? 1.0f / norm : 0.0f;

  // Packed bf16 stores: two 16-byte vectors instead of 16 scalar b16 stores.
  v8bf o0, o1;
#pragma unroll
  for (int i = 0; i < 8; ++i) {
    o0[i] = (__bf16)(buf[i] * inv);
    o1[i] = (__bf16)(buf[i + 8] * inv);
  }
  __bf16* dst = fhat + (size_t)row * F_DIM + lane * 16;
  *(v8bf*)(dst)     = o0;
  *(v8bf*)(dst + 8) = o1;

  if (lane == 0)
    recon_row[row] = dot / fmaxf(norm * sqrtf(sso), 1e-8f);
}

// ---------------------------------------------------------------------------
// Fused epilogue: fold one 16x16 sim tile into per-lane row trackers.
// All comparisons use squared angle distance (monotone), thresholds squared:
// 30^2=900, and masked candidates use the reference's +/-1e30 sentinels so
// argmax/argmin-of-all-masked yields index 0 exactly like jnp.argmax/argmin.
// Branchless: pure v_cndmask, no exec-mask divergence.
// ---------------------------------------------------------------------------
__device__ __forceinline__ void fold_tile(
    const v8f& acc, int n, int rowbase, int hw,
    const float (*sAng)[3], const int* sLab,
    const int* __restrict__ labels, const float* __restrict__ angles,
    float* bPosA, int* bPosI,
    float* bSimV, int* bSimI,
    float* bNegA, int* bNegI) {
  const int   labn = labels[n];
  const float ax = angles[n * 3 + 0];
  const float ay = angles[n * 3 + 1];
  const float az = angles[n * 3 + 2];
#pragma unroll
  for (int r = 0; r < 8; ++r) {
    const int rl  = r + 8 * hw;
    const int row = rowbase + rl;
    const float dx = sAng[rl][0] - ax;
    const float dy = sAng[rl][1] - ay;
    const float dz = sAng[rl][2] - az;
    const float ad2 = dx * dx + dy * dy + dz * dz;   // squared angle distance
    const bool diffl = (sLab[rl] != labn);
    const bool same  = (!diffl) && (row != n);

    const float pv = same ? ad2 : -1e30f;
    bool c = (pv > bPosA[r]) || (pv == bPosA[r] && n < bPosI[r]);
    bPosA[r] = c ? pv : bPosA[r];  bPosI[r] = c ? n : bPosI[r];

    const float sv = (diffl && ad2 < 900.0f) ? acc[r] : -1e30f;
    c = (sv > bSimV[r]) || (sv == bSimV[r] && n < bSimI[r]);
    bSimV[r] = c ? sv : bSimV[r];  bSimI[r] = c ? n : bSimI[r];

    const float nv = diffl ? ad2 : 1e30f;
    c = (nv < bNegA[r]) || (nv == bNegA[r] && n < bNegI[r]);
    bNegA[r] = c ? nv : bNegA[r];  bNegI[r] = c ? n : bNegI[r];
  }
}

// ---------------------------------------------------------------------------
// Kernel 2: one block per 16-row strip. 8 waves; each wave owns 32 column
// tiles, processed two at a time (A-frag reused by 2 WMMAs).
// ---------------------------------------------------------------------------
__global__ void __launch_bounds__(256)
triplet_kernel(const float* __restrict__ features,
               const int*   __restrict__ labels,
               const float* __restrict__ angles,
               const __bf16* __restrict__ fhat,
               float* __restrict__ stripSum,
               float* __restrict__ stripCnt) {
  __shared__ __align__(16) __bf16 sA[STRIP][F_DIM];   // 16 KB normalized rows
  __shared__ float sAng[STRIP][3];
  __shared__ int   sLab[STRIP];
  __shared__ float rPosA[STRIP][NWAVES]; __shared__ int rPosI[STRIP][NWAVES];
  __shared__ float rSimV[STRIP][NWAVES]; __shared__ int rSimI[STRIP][NWAVES];
  __shared__ float rNegA[STRIP][NWAVES]; __shared__ int rNegI[STRIP][NWAVES];
  __shared__ int   sPos[STRIP]; __shared__ int sNeg[STRIP];
  __shared__ float sWgt[STRIP]; __shared__ float sValid[STRIP];
  __shared__ float sContrib[STRIP];

  const int tid  = threadIdx.x;
  const int wave = tid >> 5;
  const int lane = tid & 31;
  const int hw   = lane >> 4;     // half-wave (0/1)
  const int lm   = lane & 15;
  const int rowbase = blockIdx.x * STRIP;

  // Stage this strip's normalized rows (contiguous 16x512 bf16 block).
  {
    const uint4* src = (const uint4*)(fhat + (size_t)rowbase * F_DIM);
    uint4* dst = (uint4*)(&sA[0][0]);
    for (int i = tid; i < STRIP * F_DIM * 2 / 16; i += blockDim.x) dst[i] = src[i];
  }
  if (tid < STRIP) {
    const int r = rowbase + tid;
    sAng[tid][0] = angles[r * 3 + 0];
    sAng[tid][1] = angles[r * 3 + 1];
    sAng[tid][2] = angles[r * 3 + 2];
    sLab[tid] = labels[r];
  }
  __syncthreads();

  // Per-lane trackers: 8 C-matrix rows per lane (M = r + 8*hw).
  float bPosA[8]; int bPosI[8];
  float bSimV[8]; int bSimI[8];
  float bNegA[8]; int bNegI[8];
#pragma unroll
  for (int r = 0; r < 8; ++r) {
    bPosA[r] = -1e30f; bPosI[r] = 0;
    bSimV[r] = -1e30f; bSimI[r] = 0;
    bNegA[r] =  1e30f; bNegI[r] = 0;
  }

  // Two column tiles per sweep: A-frag feeds two WMMAs.
  for (int base = 0; base < NTILES; base += 2 * NWAVES) {
    const int n0 = (base + wave) * 16 + lm;
    const int n1 = (base + NWAVES + wave) * 16 + lm;
    const __bf16* fb0 = fhat + (size_t)n0 * F_DIM;    // B = fhat^T columns
    const __bf16* fb1 = fhat + (size_t)n1 * F_DIM;

    v8f acc0 = {}, acc1 = {};
#pragma unroll 4
    for (int k0 = 0; k0 < F_DIM; k0 += 32) {
      // A frag (16x32 bf16): lane holds row lm, K = k0+16*hw..+15 contiguous.
      v8bf alo = *(const v8bf*)(&sA[lm][k0 + 16 * hw]);
      v8bf ahi = *(const v8bf*)(&sA[lm][k0 + 16 * hw + 8]);
      v16bf a = __builtin_shufflevector(alo, ahi,
                 0,1,2,3,4,5,6,7,8,9,10,11,12,13,14,15);
      // B frags (32x16 bf16): lane holds col n, K = k0+16*hw..+15 contiguous.
      v16bf b0 = *(const v16bf*)(fb0 + k0 + 16 * hw);
      v16bf b1 = *(const v16bf*)(fb1 + k0 + 16 * hw);
      acc0 = __builtin_amdgcn_wmma_f32_16x16x32_bf16(
                 false, a, false, b0, (short)0, acc0, false, false);
      acc1 = __builtin_amdgcn_wmma_f32_16x16x32_bf16(
                 false, a, false, b1, (short)0, acc1, false, false);
    }
    fold_tile(acc0, n0, rowbase, hw, sAng, sLab, labels, angles,
              bPosA, bPosI, bSimV, bSimI, bNegA, bNegI);
    fold_tile(acc1, n1, rowbase, hw, sAng, sLab, labels, angles,
              bPosA, bPosI, bSimV, bSimI, bNegA, bNegI);
  }

  // Cross-lane reduce within each 16-lane half (xor masks stay in-half).
#pragma unroll
  for (int r = 0; r < 8; ++r) {
#pragma unroll
    for (int m = 8; m >= 1; m >>= 1) {
      float ov; int oi; bool c;
      ov = __shfl_xor(bPosA[r], m, 32); oi = __shfl_xor(bPosI[r], m, 32);
      c = (ov > bPosA[r]) || (ov == bPosA[r] && oi < bPosI[r]);
      bPosA[r] = c ? ov : bPosA[r]; bPosI[r] = c ? oi : bPosI[r];
      ov = __shfl_xor(bSimV[r], m, 32); oi = __shfl_xor(bSimI[r], m, 32);
      c = (ov > bSimV[r]) || (ov == bSimV[r] && oi < bSimI[r]);
      bSimV[r] = c ? ov : bSimV[r]; bSimI[r] = c ? oi : bSimI[r];
      ov = __shfl_xor(bNegA[r], m, 32); oi = __shfl_xor(bNegI[r], m, 32);
      c = (ov < bNegA[r]) || (ov == bNegA[r] && oi < bNegI[r]);
      bNegA[r] = c ? ov : bNegA[r]; bNegI[r] = c ? oi : bNegI[r];
    }
    if (lm == 0) {
      const int rl = r + 8 * hw;
      rPosA[rl][wave] = bPosA[r]; rPosI[rl][wave] = bPosI[r];
      rSimV[rl][wave] = bSimV[r]; rSimI[rl][wave] = bSimI[r];
      rNegA[rl][wave] = bNegA[r]; rNegI[rl][wave] = bNegI[r];
    }
  }
  __syncthreads();

  // Per-row final combine across waves (fixed order -> deterministic).
  if (tid < STRIP) {
    float pA = -1e30f; int pI = 0;
    float sV = -1e30f; int sI = 0;
    float nA =  1e30f; int nI = 0;
    for (int w = 0; w < NWAVES; ++w) {
      float v; int i; bool c;
      v = rPosA[tid][w]; i = rPosI[tid][w];
      c = (v > pA) || (v == pA && i < pI); pA = c ? v : pA; pI = c ? i : pI;
      v = rSimV[tid][w]; i = rSimI[tid][w];
      c = (v > sV) || (v == sV && i < sI); sV = c ? v : sV; sI = c ? i : sI;
      v = rNegA[tid][w]; i = rNegI[tid][w];
      c = (v < nA) || (v == nA && i < nI); nA = c ? v : nA; nI = c ? i : nI;
    }
    const bool has_pos = pA > -1e29f;
    const bool has_neg = nA <  1e29f;
    const bool sim_any = sV > -1e29f;
    const int neg = sim_any ? sI : nI;

    // Squared-distance weight thresholds: 45^2=2025, 15^2=225.
    float dx = angles[pI * 3 + 0] - sAng[tid][0];
    float dy = angles[pI * 3 + 1] - sAng[tid][1];
    float dz = angles[pI * 3 + 2] - sAng[tid][2];
    const float pa2 = dx * dx + dy * dy + dz * dz;
    dx = angles[neg * 3 + 0] - sAng[tid][0];
    dy = angles[neg * 3 + 1] - sAng[tid][1];
    dz = angles[neg * 3 + 2] - sAng[tid][2];
    const float na2 = dx * dx + dy * dy + dz * dz;

    sPos[tid] = pI;
    sNeg[tid] = neg;
    sWgt[tid] = (pa2 > 2025.0f ? 2.0f : 1.0f) * (na2 < 225.0f ? 1.5f : 1.0f);
    sValid[tid] = (has_pos && has_neg) ? 1.0f : 0.0f;
  }
  __syncthreads();

  // Triplet distances on RAW features (eps added per-component, as reference).
  for (int rr = wave * 2; rr < wave * 2 + 2; ++rr) {
    const int row = rowbase + rr;
    const float* fa = features + (size_t)row       * F_DIM + lane * 16;
    const float* fp = features + (size_t)sPos[rr]  * F_DIM + lane * 16;
    const float* fn = features + (size_t)sNeg[rr]  * F_DIM + lane * 16;
    float sp = 0.f, sn = 0.f;
#pragma unroll
    for (int i = 0; i < 16; ++i) {
      const float a  = fa[i];
      const float dp = a - fp[i] + 1e-6f; sp += dp * dp;
      const float dn = a - fn[i] + 1e-6f; sn += dn * dn;
    }
    sp = wave_reduce_add(sp);
    sn = wave_reduce_add(sn);
    if (lane == 0) {
      const float basic = fmaxf(sqrtf(sp) - sqrtf(sn) + 0.2f, 0.0f);
      sContrib[rr] = sWgt[rr] * basic * sValid[rr];
    }
  }
  __syncthreads();

  if (tid == 0) {
    float s = 0.f, c = 0.f;
    for (int r = 0; r < STRIP; ++r) { s += sContrib[r]; c += sValid[r]; }
    stripSum[blockIdx.x] = s;
    stripCnt[blockIdx.x] = c;
  }
}

// ---------------------------------------------------------------------------
// Kernel 3: deterministic final reduction -> scalar loss.
// ---------------------------------------------------------------------------
__global__ void final_kernel(const float* __restrict__ stripSum,
                             const float* __restrict__ stripCnt,
                             const float* __restrict__ recon_row,
                             float* __restrict__ out,
                             int nstrips, int b) {
  __shared__ float s1[256], s2[256], s3[256];
  const int tid = threadIdx.x;
  float a = 0.f, c = 0.f, r = 0.f;
  for (int i = tid; i < nstrips; i += 256) { a += stripSum[i]; c += stripCnt[i]; }
  for (int i = tid; i < b; i += 256) r += recon_row[i];
  s1[tid] = a; s2[tid] = c; s3[tid] = r;
  __syncthreads();
  for (int st = 128; st >= 1; st >>= 1) {
    if (tid < st) { s1[tid] += s1[tid + st]; s2[tid] += s2[tid + st]; s3[tid] += s3[tid + st]; }
    __syncthreads();
  }
  if (tid == 0) {
    const float cnt = fmaxf(s2[0], 1.0f);
    out[0] = s1[0] / cnt + 0.1f * (1.0f - s3[0] / (float)b);
  }
}

// ---------------------------------------------------------------------------
extern "C" void kernel_launch(void* const* d_in, const int* in_sizes, int n_in,
                              void* d_out, int out_size, void* d_ws, size_t ws_size,
                              hipStream_t stream) {
  const float* features = (const float*)d_in[0];   // [4096,512] f32
  const int*   labels   = (const int*)  d_in[1];   // [4096]     i32
  const float* angles   = (const float*)d_in[2];   // [4096,3]   f32
  const float* forig    = (const float*)d_in[3];   // [4096,512] f32
  float* out = (float*)d_out;

  char* ws = (char*)d_ws;
  __bf16* fhat      = (__bf16*)ws;                               // 4 MB
  float*  recon_row = (float*)(ws + (size_t)B_DIM * F_DIM * 2);  // 16 KB
  float*  stripSum  = recon_row + B_DIM;                         // 1 KB
  float*  stripCnt  = stripSum + (B_DIM / STRIP);                // 1 KB

  prep_kernel<<<B_DIM / NWAVES, 256, 0, stream>>>(features, forig, fhat, recon_row);
  triplet_kernel<<<B_DIM / STRIP, 256, 0, stream>>>(features, labels, angles,
                                                    fhat, stripSum, stripCnt);
  final_kernel<<<1, 256, 0, stream>>>(stripSum, stripCnt, recon_row, out,
                                      B_DIM / STRIP, B_DIM);
}
